// FloodGNN_78159814852851
// MI455X (gfx1250) — compile-verified, matrix operand order checked by reading, not run on hardware
//
#include <hip/hip_runtime.h>
#include <hip/hip_bf16.h>

// ---------------------------------------------------------------------------
// FloodGNN (3-layer GAT, 4 heads) forward for MI455X (gfx1250).
//
// Roofline reasoning:
//  * Dense GEMMs (h@W): ~14.7 GFLOP -> WMMA f16->f32 (v_wmma_f32_16x16x32_f16).
//  * Edge softmax + scatter: xs (51.2MB) and agg (51.2MB) both fit in the
//    192MB L2, so the gather/atomic-scatter phase runs out of L2, not HBM.
//  * wave32 everywhere: reductions use __shfl_xor(.,m,32).
//  * GEMM staging is double-buffered in LDS: one workgroup barrier per K-step,
//    vectorized f32x2 loads + packed f16x2 LDS stores.
// ---------------------------------------------------------------------------

typedef _Float16 v2h  __attribute__((ext_vector_type(2)));
typedef _Float16 v8h  __attribute__((ext_vector_type(8)));
typedef _Float16 v16h __attribute__((ext_vector_type(16)));
typedef float    v8f  __attribute__((ext_vector_type(8)));

namespace {
constexpr int    kN    = 50000;
constexpr int    kE    = 800000;
constexpr int    kEtot = kE + kN;      // with self loops
constexpr int    kIN   = 16;
constexpr int    kHID  = 64;
constexpr int    kHEADS= 4;
constexpr int    kDIM  = kHID * kHEADS;  // 256
constexpr float  kNegSlope = 0.2f;
constexpr int    kAStride  = 40;       // LDS halves stride (16B aligned, bank-staggered)
}

// ---------------- small helpers ----------------
__device__ __forceinline__ float waveSum32(float v) {
  #pragma unroll
  for (int m = 16; m >= 1; m >>= 1) v += __shfl_xor(v, m, 32);
  return v;
}

__device__ __forceinline__ void atomicMaxF(float* addr, float v) {
  // standard bit trick: monotone mapping of float order onto int/uint order
  if (v >= 0.0f) atomicMax(reinterpret_cast<int*>(addr), __float_as_int(v));
  else           atomicMin(reinterpret_cast<unsigned int*>(addr), __float_as_uint(v));
}

__global__ void fill_kernel(float* __restrict__ p, float v, size_t n) {
  size_t i = (size_t)blockIdx.x * blockDim.x + threadIdx.x;
  if (i < n) p[i] = v;
}

// ---------------- stage 0: input projection [N,16]@[16,64]+b, relu --------
__global__ void proj_kernel(const float* __restrict__ x,
                            const float* __restrict__ W,
                            const float* __restrict__ b,
                            float* __restrict__ h) {
  int idx = blockIdx.x * blockDim.x + threadIdx.x;   // N*64
  if (idx >= kN * kHID) return;
  int n = idx >> 6, j = idx & 63;
  float s = b[j];
  #pragma unroll
  for (int k = 0; k < kIN; ++k) s = fmaf(x[n * kIN + k], W[k * kHID + j], s);
  h[idx] = fmaxf(s, 0.0f);
}

// ---------------- W [K,256] f32 -> Wt [256,K] f16 (B operand, N-major) ----
__global__ void wt_f16_kernel(const float* __restrict__ W,
                              _Float16* __restrict__ Wt, int K) {
  int idx = blockIdx.x * blockDim.x + threadIdx.x;   // K*256
  if (idx >= K * kDIM) return;
  int k = idx / kDIM, n = idx % kDIM;
  Wt[(size_t)n * K + k] = (_Float16)W[idx];
}

// ---------------- WMMA GEMM: C[N,256] = A[N,K] @ W[K,256] -----------------
// grid = (2, N/16); block = 256 (8 waves). Wave w covers cols
// [blockIdx.x*128 + w*16, +16). The shared 16x32 f16 A tile is staged through
// a double-buffered LDS region (one barrier per K-step); B fragments come
// straight from the transposed f16 weights (L2 resident, 131KB).
__global__ __launch_bounds__(256) void gemm_xs_wmma(
    const float* __restrict__ A, const _Float16* __restrict__ Bt,
    float* __restrict__ C, int K) {
  __shared__ _Float16 As[2][16 * kAStride];

  const int lane = threadIdx.x & 31;
  const int wave = threadIdx.x >> 5;
  const int m0 = blockIdx.y * 16;
  const int colBase = blockIdx.x * 128 + wave * 16;

  // cooperative loader mapping: thread -> (row, even K pair)
  const int lrow = threadIdx.x >> 4;          // 0..15
  const int lk   = (threadIdx.x & 15) * 2;    // 0,2,...,30
  const float* aRow = A + (size_t)(m0 + lrow) * K + lk;

  const int row = lane & 15;                  // A-frag row / B-frag col (ISA layout)
  const int kb  = (lane >> 4) ? 8 : 0;        // K-group base by lane half
  const _Float16* bcolp = Bt + (size_t)(colBase + row) * K;

  auto stage = [&](int buf, int k0) {
    float2 f = *reinterpret_cast<const float2*>(aRow + k0);   // global_load_b64
    v2h p2; p2[0] = (_Float16)f.x; p2[1] = (_Float16)f.y;
    *reinterpret_cast<v2h*>(&As[buf][lrow * kAStride + lk]) = p2;  // ds_store_b32
  };

  stage(0, 0);
  v8f acc = {};
  int p = 0;
  for (int k0 = 0; k0 < K; k0 += 32) {
    __syncthreads();                          // buf[p] tile ready for all waves
    if (k0 + 32 < K) stage(p ^ 1, k0 + 32);   // overlap next-tile staging

    v16h a, b;
    {
      const _Float16* ap = &As[p][row * kAStride + kb];
      v8h lo = *reinterpret_cast<const v8h*>(ap);
      v8h hi = *reinterpret_cast<const v8h*>(ap + 16);
      #pragma unroll
      for (int i = 0; i < 8; ++i) { a[i] = lo[i]; a[8 + i] = hi[i]; }
    }
    {
      const _Float16* bp = bcolp + k0 + kb;
      v8h lo = *reinterpret_cast<const v8h*>(bp);
      v8h hi = *reinterpret_cast<const v8h*>(bp + 16);
      #pragma unroll
      for (int i = 0; i < 8; ++i) { b[i] = lo[i]; b[8 + i] = hi[i]; }
    }
    acc = __builtin_amdgcn_wmma_f32_16x16x32_f16(
        /*neg_a=*/false, a, /*neg_b=*/false, b,
        /*c_mod=*/(short)0, acc, /*reuse_a=*/false, /*reuse_b=*/false);
    p ^= 1;
  }

  // C/D layout: lanes 0-15 -> N=lane, M=r; lanes 16-31 -> N=lane-16, M=r+8
  const int ccol = colBase + (lane & 15);
  const int radd = (lane >> 4) << 3;
  #pragma unroll
  for (int r = 0; r < 8; ++r)
    C[(size_t)(m0 + r + radd) * kDIM + ccol] = acc[r];
}

// ---------------- attention scores s_src/s_dst [N,4] ----------------------
__global__ void attn_scores(const float* __restrict__ xs,
                            const float* __restrict__ asrc,
                            const float* __restrict__ adst,
                            float* __restrict__ ssrc,
                            float* __restrict__ sdst) {
  int idx = blockIdx.x * blockDim.x + threadIdx.x;   // N*4
  if (idx >= kN * kHEADS) return;
  int n = idx >> 2, hh = idx & 3;
  const float* xp = xs + (size_t)n * kDIM + hh * kHID;
  const float* ap = asrc + hh * kHID;
  const float* bp = adst + hh * kHID;
  float s1 = 0.f, s2 = 0.f;
  #pragma unroll 4
  for (int k = 0; k < kHID; ++k) { float v = xp[k]; s1 = fmaf(v, ap[k], s1); s2 = fmaf(v, bp[k], s2); }
  ssrc[idx] = s1; sdst[idx] = s2;
}

__device__ __forceinline__ void edge_endpoints(const int* __restrict__ ei,
                                               int e, int& s, int& d) {
  if (e < kE) { s = ei[e]; d = ei[kE + e]; }
  else        { s = e - kE; d = e - kE; }     // self loop
}

// ---------------- segment max of leaky-relu scores ------------------------
__global__ void edge_max_kernel(const int* __restrict__ ei,
                                const float* __restrict__ ssrc,
                                const float* __restrict__ sdst,
                                float* __restrict__ amax) {
  int e = blockIdx.x * blockDim.x + threadIdx.x;
  if (e >= kEtot) return;
  int s, d; edge_endpoints(ei, e, s, d);
  #pragma unroll
  for (int hh = 0; hh < kHEADS; ++hh) {
    float a = ssrc[s * kHEADS + hh] + sdst[d * kHEADS + hh];
    a = a > 0.f ? a : kNegSlope * a;
    atomicMaxF(&amax[d * kHEADS + hh], a);
  }
}

// ---------------- segment sum of exp(score - max) -------------------------
__global__ void edge_sum_kernel(const int* __restrict__ ei,
                                const float* __restrict__ ssrc,
                                const float* __restrict__ sdst,
                                const float* __restrict__ amax,
                                float* __restrict__ den) {
  int e = blockIdx.x * blockDim.x + threadIdx.x;
  if (e >= kEtot) return;
  int s, d; edge_endpoints(ei, e, s, d);
  #pragma unroll
  for (int hh = 0; hh < kHEADS; ++hh) {
    float a = ssrc[s * kHEADS + hh] + sdst[d * kHEADS + hh];
    a = a > 0.f ? a : kNegSlope * a;
    atomicAdd(&den[d * kHEADS + hh], expf(a - amax[d * kHEADS + hh]));
  }
}

// ---------------- weighted message scatter (one wave per edge) ------------
__global__ __launch_bounds__(256) void edge_msg_kernel(
    const int* __restrict__ ei, const float* __restrict__ xs,
    const float* __restrict__ ssrc, const float* __restrict__ sdst,
    const float* __restrict__ amax, const float* __restrict__ den,
    float* __restrict__ agg) {
  const int lane = threadIdx.x & 31;
  const int e = blockIdx.x * 8 + (threadIdx.x >> 5);
  if (e >= kEtot) return;
  int s, d; edge_endpoints(ei, e, s, d);

  const float* xp = xs + (size_t)s * kDIM;
  __builtin_prefetch(xp + lane * 8, 0, 1);    // warm the gathered xs row (L2)

  float w[kHEADS];
  #pragma unroll
  for (int hh = 0; hh < kHEADS; ++hh) {
    float a = ssrc[s * kHEADS + hh] + sdst[d * kHEADS + hh];
    a = a > 0.f ? a : kNegSlope * a;
    w[hh] = expf(a - amax[d * kHEADS + hh]) / (den[d * kHEADS + hh] + 1e-16f);
  }
  float* op = agg + (size_t)d * kDIM;
  #pragma unroll
  for (int i = 0; i < 8; ++i) {
    int c = lane + 32 * i;                 // c>>6 == i>>1 : constant head index
    atomicAdd(&op[c], xp[c] * w[i >> 1]);
  }
}

// ---------------- bias + LayerNorm + residual + relu (in place on agg) ----
__global__ __launch_bounds__(256) void post_kernel(
    float* __restrict__ agg, const float* __restrict__ hin,
    const float* __restrict__ bias, const float* __restrict__ g,
    const float* __restrict__ bvec, int has_res) {
  const int lane = threadIdx.x & 31;
  const int n = blockIdx.x * 8 + (threadIdx.x >> 5);
  if (n >= kN) return;
  float v[8];
  float s = 0.f;
  #pragma unroll
  for (int i = 0; i < 8; ++i) {
    int c = lane + 32 * i;
    v[i] = agg[(size_t)n * kDIM + c] + bias[c];
    s += v[i];
  }
  float mu = waveSum32(s) * (1.0f / kDIM);
  float vs = 0.f;
  #pragma unroll
  for (int i = 0; i < 8; ++i) { float dv = v[i] - mu; vs += dv * dv; }
  float var = waveSum32(vs) * (1.0f / kDIM);
  float rs = rsqrtf(var + 1e-5f);
  #pragma unroll
  for (int i = 0; i < 8; ++i) {
    int c = lane + 32 * i;
    float y = (v[i] - mu) * rs * g[c] + bvec[c];
    if (has_res) y += hin[(size_t)n * kDIM + c];
    agg[(size_t)n * kDIM + c] = fmaxf(y, 0.f);
  }
}

// ---------------- MLP head: sigmoid(relu(h@W1+b1)@W2+b2) ------------------
__global__ __launch_bounds__(256) void heads_kernel(
    const float* __restrict__ h, const float* __restrict__ W1,
    const float* __restrict__ b1, const float* __restrict__ W2,
    const float* __restrict__ b2, float* __restrict__ out) {
  const int lane = threadIdx.x & 31;                  // hidden unit j (32 of them)
  const int n = blockIdx.x * 8 + (threadIdx.x >> 5);
  if (n >= kN) return;
  const float* hp = h + (size_t)n * kDIM;
  float t = b1[lane];
  #pragma unroll 4
  for (int k = 0; k < kDIM; ++k) t = fmaf(hp[k], W1[k * 32 + lane], t);
  t = fmaxf(t, 0.f);
  float z = waveSum32(t * W2[lane]);
  if (lane == 0) out[n] = 1.0f / (1.0f + expf(-(z + b2[0])));
}

// ---------------------------------------------------------------------------
extern "C" void kernel_launch(void* const* d_in, const int* in_sizes, int n_in,
                              void* d_out, int out_size, void* d_ws, size_t ws_size,
                              hipStream_t stream) {
  (void)in_sizes; (void)n_in; (void)out_size; (void)ws_size;

  // ---- inputs (setup_inputs order, lists flattened) ----
  const float* x       = (const float*)d_in[0];
  const int*   ei      = (const int*)  d_in[1];
  const float* proj_W  = (const float*)d_in[2];
  const float* proj_b  = (const float*)d_in[3];
  const float* gat_lin[3]  = {(const float*)d_in[4],  (const float*)d_in[5],  (const float*)d_in[6]};
  const float* att_src[3]  = {(const float*)d_in[7],  (const float*)d_in[8],  (const float*)d_in[9]};
  const float* att_dst[3]  = {(const float*)d_in[10], (const float*)d_in[11], (const float*)d_in[12]};
  const float* gat_bias[3] = {(const float*)d_in[13], (const float*)d_in[14], (const float*)d_in[15]};
  const float* norm_g[3]   = {(const float*)d_in[16], (const float*)d_in[17], (const float*)d_in[18]};
  const float* norm_b[3]   = {(const float*)d_in[19], (const float*)d_in[20], (const float*)d_in[21]};
  // heads_p: 4 x (W1, b1, W2, b2) at indices 22..37
  float* out = (float*)d_out;

  // ---- workspace layout ----
  const size_t nodeF = (size_t)kN * kDIM;              // 12.8M floats
  char* wsp = (char*)d_ws;
  float*    hA   = (float*)wsp;                   wsp += nodeF * sizeof(float);
  float*    hB   = (float*)wsp;                   wsp += nodeF * sizeof(float);
  float*    xs   = (float*)wsp;                   wsp += nodeF * sizeof(float);
  _Float16* Wt   = (_Float16*)wsp;                wsp += (size_t)kDIM * kDIM * sizeof(_Float16);
  float*    ssrc = (float*)wsp;                   wsp += (size_t)kN * kHEADS * sizeof(float);
  float*    sdst = (float*)wsp;                   wsp += (size_t)kN * kHEADS * sizeof(float);
  float*    amax = (float*)wsp;                   wsp += (size_t)kN * kHEADS * sizeof(float);
  float*    den  = (float*)wsp;                   wsp += (size_t)kN * kHEADS * sizeof(float);

  const int TB = 256;
  auto blocks = [](size_t n, int tb) { return (unsigned)((n + tb - 1) / tb); };

  // 0) input projection -> hA [N,64]
  proj_kernel<<<blocks((size_t)kN * kHID, TB), TB, 0, stream>>>(x, proj_W, proj_b, hA);

  float* cur = hA;
  float* nxt = hB;
  int K = kHID;

  for (int l = 0; l < 3; ++l) {
    // weights -> transposed f16
    wt_f16_kernel<<<blocks((size_t)K * kDIM, TB), TB, 0, stream>>>(gat_lin[l], Wt, K);

    // xs = cur @ W  (WMMA)
    dim3 ggrid(2, kN / 16);
    gemm_xs_wmma<<<ggrid, TB, 0, stream>>>(cur, Wt, xs, K);

    // attention scores
    attn_scores<<<blocks((size_t)kN * kHEADS, TB), TB, 0, stream>>>(
        xs, att_src[l], att_dst[l], ssrc, sdst);

    // init reductions + aggregation buffer
    fill_kernel<<<blocks((size_t)kN * kHEADS, TB), TB, 0, stream>>>(amax, -3.0e38f, (size_t)kN * kHEADS);
    fill_kernel<<<blocks((size_t)kN * kHEADS, TB), TB, 0, stream>>>(den, 0.0f, (size_t)kN * kHEADS);
    fill_kernel<<<blocks(nodeF, TB), TB, 0, stream>>>(nxt, 0.0f, nodeF);

    // segment softmax + scatter
    edge_max_kernel<<<blocks((size_t)kEtot, TB), TB, 0, stream>>>(ei, ssrc, sdst, amax);
    edge_sum_kernel<<<blocks((size_t)kEtot, TB), TB, 0, stream>>>(ei, ssrc, sdst, amax, den);
    edge_msg_kernel<<<blocks((size_t)kEtot * 32, TB), TB, 0, stream>>>(
        ei, xs, ssrc, sdst, amax, den, nxt);

    // bias + LN + residual + relu (residual only when dims match: l>0)
    post_kernel<<<blocks((size_t)kN * 32, TB), TB, 0, stream>>>(
        nxt, cur, gat_bias[l], norm_g[l], norm_b[l], l > 0 ? 1 : 0);

    float* tmp = cur; cur = nxt; nxt = tmp;
    K = kDIM;
  }

  // 4 output heads -> out[head*N + n]
  for (int t = 0; t < 4; ++t) {
    heads_kernel<<<blocks((size_t)kN * 32, TB), TB, 0, stream>>>(
        cur,
        (const float*)d_in[22 + 4 * t], (const float*)d_in[23 + 4 * t],
        (const float*)d_in[24 + 4 * t], (const float*)d_in[25 + 4 * t],
        out + (size_t)t * kN);
  }
}
